// RealAgnosticAttResidualInteractionBlock_28939489641130
// MI455X (gfx1250) — matrix operand order repaired and checked by reading, not compiled, not run on hardware
//
#include <hip/hip_runtime.h>

typedef __attribute__((ext_vector_type(16))) __bf16 v16bf;
typedef __attribute__((ext_vector_type(8)))  float  v8f;

union __align__(16) Frag {
  v16bf v;
  unsigned short s[16];
  uint4 q[2];
};

__device__ __forceinline__ unsigned short f2bf(float x) {
  unsigned int u = __float_as_uint(x);
  u += 0x7FFFu + ((u >> 16) & 1u);   // round-to-nearest-even
  return (unsigned short)(u >> 16);
}

__device__ __forceinline__ float bf2f(unsigned short b) {
  return __uint_as_float(((unsigned int)b) << 16);
}

// silu via hardware reciprocal (v_rcp_f32): avoids the IEEE divide expansion
// (div_scale/div_fmas chain) that serializes the VALU pipe after each WMMA.
__device__ __forceinline__ float silu_f(float x) {
  return x * __builtin_amdgcn_rcpf(1.0f + __expf(-x));
}

// Async global->LDS copy (CDNA5): 16B per lane, tracked by ASYNCcnt.
__device__ __forceinline__ void async_copy_b128(unsigned lds_off, const void* gptr) {
  asm volatile("global_load_async_to_lds_b128 %0, %1, off"
               :: "v"(lds_off), "v"((unsigned long long)(size_t)gptr)
               : "memory");
}
__device__ __forceinline__ void wait_async0() {
  asm volatile("s_wait_asynccnt 0" ::: "memory");
}

// ---------------------------------------------------------------------------
// Pack an f32 weight matrix W[K x Ncols] (row-major) into bf16 WMMA-B-fragment
// order, scale folded in. Fragment (kt,nt): lane l holds column n = nt*16+(l&15),
// k = kt*32 + 8*(l>>4) + (j&7) + 16*(j>>3) for j=0..15 (16 contiguous bf16 per lane = 32B).
// ---------------------------------------------------------------------------
__global__ void pack_weights(const float* __restrict__ W, int K, int Ncols,
                             unsigned short* __restrict__ Bp,
                             int Ktiles, int Ntiles, float scale) {
  int idx = blockIdx.x * blockDim.x + threadIdx.x;
  int total = Ktiles * Ntiles * 32;
  if (idx >= total) return;
  int lane = idx & 31;
  int frag = idx >> 5;
  int nt = frag % Ntiles;
  int kt = frag / Ntiles;
  int n = nt * 16 + (lane & 15);
  int kbase = kt * 32 + (lane >> 4) * 8;
  __align__(16) unsigned short out[16];
#pragma unroll
  for (int j = 0; j < 16; ++j) {
    int k = kbase + (j & 7) + ((j >> 3) << 4);
    float v = (k < K && n < Ncols) ? W[(long)k * Ncols + n] * scale : 0.0f;
    out[j] = f2bf(v);
  }
  uint4* dst = (uint4*)(Bp + (long)idx * 16);
  dst[0] = ((uint4*)out)[0];
  dst[1] = ((uint4*)out)[1];
}

__global__ void zero_f32(float* __restrict__ p, long n) {
  long i = (long)blockIdx.x * blockDim.x + threadIdx.x;
  long stride = (long)gridDim.x * blockDim.x;
  for (; i < n; i += stride) p[i] = 0.0f;
}

// ---------------------------------------------------------------------------
// Generic strided bf16-WMMA GEMM: Out[m,n] = sum_k A[m,k] * Bp[k,n] * outScale
// A is f32 with arbitrary row/k element strides (handles the (N,C,3) einsums).
// One wave per block; block = one 16-row M-tile; loops all N-tiles, A frags
// loaded once and reused.
// ---------------------------------------------------------------------------
template <int KT, int NT>
__global__ __launch_bounds__(32) void wmma_gemm(
    const float* __restrict__ A, int ldaRow, int ldaK,
    const unsigned short* __restrict__ Bp,
    float* __restrict__ Out, int ldoRow, int ldoCol, float outScale) {
  int lane = threadIdx.x;
  int mtile = blockIdx.x;
  int m = mtile * 16 + (lane & 15);
  int khalf = lane >> 4;

  Frag af[KT];
#pragma unroll
  for (int kt = 0; kt < KT; ++kt) {
#pragma unroll
    for (int j = 0; j < 16; ++j) {
      int k = kt * 32 + khalf * 8 + (j & 7) + ((j >> 3) << 4);
      af[kt].s[j] = f2bf(A[(long)m * ldaRow + (long)k * ldaK]);
    }
  }

#pragma unroll
  for (int nt = 0; nt < NT; ++nt) {
    v8f acc = {};
#pragma unroll
    for (int kt = 0; kt < KT; ++kt) {
      Frag bf_;
      const uint4* pb = (const uint4*)(Bp + (((long)kt * NT + nt) * 32 + lane) * 16);
      bf_.q[0] = pb[0];
      bf_.q[1] = pb[1];
      acc = __builtin_amdgcn_wmma_f32_16x16x32_bf16(
          false, af[kt].v, false, bf_.v, (short)0, acc, false, false);
    }
    int n = nt * 16 + (lane & 15);
    int mrow = mtile * 16 + khalf * 8;
#pragma unroll
    for (int r = 0; r < 8; ++r) {
      Out[(long)(mrow + r) * ldoRow + (long)n * ldoCol] = acc[r] * outScale;
    }
  }
}

// ---------------------------------------------------------------------------
// Edge-MLP layer over a 64-edge tile (4 M-tiles of 16). Waves split N-tiles;
// B fragments of one N-tile are held in registers and reused across the 4
// M-tiles (4x cut in L2 weight traffic); A fragments re-read from LDS (cheap).
// ---------------------------------------------------------------------------
template <int KT, int NT, bool SILU>
__device__ __forceinline__ void mlp_layer64(
    const unsigned short* Asrc, int lda,                 // bf16 LDS, 64 rows
    const unsigned short* __restrict__ Bp,               // packed global bf16
    unsigned short* Hout, int ldo,                       // bf16 LDS out
    int wave, int lane) {
  const int m = lane & 15;
  const int khalf = lane >> 4;
#pragma unroll
  for (int nt0 = 0; nt0 < NT / 4; ++nt0) {
    int nt = nt0 * 4 + wave;
    Frag bfr[KT];
#pragma unroll
    for (int kt = 0; kt < KT; ++kt) {
      const uint4* pb = (const uint4*)(Bp + (((long)kt * NT + nt) * 32 + lane) * 16);
      bfr[kt].q[0] = pb[0];
      bfr[kt].q[1] = pb[1];
    }
#pragma unroll
    for (int mt = 0; mt < 4; ++mt) {
      v8f acc = {};
#pragma unroll
      for (int kt = 0; kt < KT; ++kt) {
        Frag af;
        const uint4* pa =
            (const uint4*)(Asrc + (mt * 16 + m) * lda + kt * 32 + khalf * 8);
        af.q[0] = pa[0];   // k = base .. base+7
        af.q[1] = pa[2];   // k = base+16 .. base+23
        acc = __builtin_amdgcn_wmma_f32_16x16x32_bf16(
            false, af.v, false, bfr[kt].v, (short)0, acc, false, false);
      }
      int n = nt * 16 + m;
#pragma unroll
      for (int r = 0; r < 8; ++r) {
        float v = acc[r];
        if (SILU) v = silu_f(v);
        Hout[(mt * 16 + khalf * 8 + r) * ldo + n] = f2bf(v);
      }
    }
  }
}

// ---------------------------------------------------------------------------
// Fused edge pipeline, 64 edges / block, 128 threads (4 waves):
//   async-gather aug(f32) -> bf16 -> 4-layer bf16-WMMA MLP (LDS-resident) ->
//   tensor product -> atomic scatter into m0 (N,256), m1 (N,3,256).
// Dynamic LDS layout (136 KB, aliased by liveness):
//   [0,      40960) : augF f32 (64x160)   ... later reused as sHb (32 KB)
//   [40960,  73728) : sHa bf16 (64x256)
//   [73728, 139264) : augB bf16 (64x160) during L1, then sT bf16 (64x512)
// ---------------------------------------------------------------------------
__global__ __launch_bounds__(128) void edge_mlp_scatter(
    const float* __restrict__ edge_feats,   // (E,8)
    const float* __restrict__ edge_attrs,   // (E,4)
    const int* __restrict__ edge_index,     // (2,E)
    int E,
    const float* __restrict__ down,         // (N,64)
    const float* __restrict__ u0,           // (N,128)
    const float* __restrict__ u1,           // (N,3,128)
    const unsigned short* __restrict__ w1p,
    const unsigned short* __restrict__ w2p,
    const unsigned short* __restrict__ w3p,
    const unsigned short* __restrict__ w4p,
    float* __restrict__ m0,                 // (N,256)
    float* __restrict__ m1) {               // (N,3,256)
  extern __shared__ char smem[];
  float*          augF = (float*)smem;                              // 40960 B
  unsigned short* sHb  = (unsigned short*)smem;                     // alias
  unsigned short* sHa  = (unsigned short*)(smem + 40960);           // 32768 B
  unsigned short* augB = (unsigned short*)(smem + 40960 + 32768);   // alias of sT
  unsigned short* sT   = (unsigned short*)(smem + 40960 + 32768);   // 65536 B

  const int tid = threadIdx.x;
  const int wave = tid >> 5;
  const int lane = tid & 31;
  const int e0 = blockIdx.x * 64;

  __builtin_prefetch(w1p, 0, 0);   // global_prefetch_b8: warm first weight layer
  __builtin_prefetch(w2p, 0, 0);

  // Phase 0: async-gather aug f32 = [edge_feats(8) | down[send](64) | down[recv](64)]
  // 34 quad-float (16B) async copies per edge, no VGPR round-trip.
  for (int q = tid; q < 64 * 34; q += 128) {
    int e = q / 34;
    int j = q - e * 34;
    int ge = e0 + e;
    const float* src;
    int col4;
    if (j < 2) {
      src = edge_feats + (long)ge * 8 + j * 4;
      col4 = j * 4;
    } else if (j < 18) {
      int s = edge_index[ge];
      src = down + (long)s * 64 + (j - 2) * 4;
      col4 = 8 + (j - 2) * 4;
    } else {
      int r = edge_index[E + ge];
      src = down + (long)r * 64 + (j - 18) * 4;
      col4 = 72 + (j - 18) * 4;
    }
    unsigned lds_off = (unsigned)(size_t)(augF + e * 160 + col4);
    async_copy_b128(lds_off, src);
  }
  wait_async0();
  __syncthreads();

  // Phase 1: convert aug f32 -> bf16 (zero-pad cols 136..159 for K=160).
  for (int slot = tid; slot < 64 * 160; slot += 128) {
    int col = slot % 160;
    augB[slot] = (col < 136) ? f2bf(augF[slot]) : (unsigned short)0;
  }
  __syncthreads();

  // Layer 1: 160 -> 256, silu   (augB -> sHa; augF becomes dead)
  mlp_layer64<5, 16, true>(augB, 160, w1p, sHa, 256, wave, lane);
  __syncthreads();
  // Layer 2: 256 -> 256, silu   (sHa -> sHb, which aliases dead augF)
  mlp_layer64<8, 16, true>(sHa, 256, w2p, sHb, 256, wave, lane);
  __syncthreads();
  // Layer 3: 256 -> 256, silu   (sHb -> sHa)
  mlp_layer64<8, 16, true>(sHb, 256, w3p, sHa, 256, wave, lane);
  __syncthreads();
  // Layer 4: 256 -> 512, linear (sHa -> sT, which aliases dead augB)
  mlp_layer64<8, 32, false>(sHa, 256, w4p, sT, 512, wave, lane);
  __syncthreads();

  // Tensor product + scatter. Thread t handles channel c = t for each edge.
  const int c = tid;  // 0..127
  const float inv_sqrt3 = 0.57735026918962576f;
#pragma unroll 1
  for (int e = 0; e < 64; ++e) {
    int ge = e0 + e;
    int s = edge_index[ge];
    int rcv = edge_index[E + ge];
    float w0 = bf2f(sT[e * 512 + c]);
    float w1 = bf2f(sT[e * 512 + 128 + c]);
    float w2 = bf2f(sT[e * 512 + 256 + c]);
    float w3 = bf2f(sT[e * 512 + 384 + c]);
    float xs0  = u0[(long)s * 128 + c];
    float xs1a = u1[(long)s * 384 + 0 * 128 + c];
    float xs1b = u1[(long)s * 384 + 1 * 128 + c];
    float xs1c = u1[(long)s * 384 + 2 * 128 + c];
    float y0  = edge_attrs[(long)ge * 4 + 0];
    float y1a = edge_attrs[(long)ge * 4 + 1];
    float y1b = edge_attrs[(long)ge * 4 + 2];
    float y1c = edge_attrs[(long)ge * 4 + 3];
    float dot = xs1a * y1a + xs1b * y1b + xs1c * y1c;

    atomicAdd(&m0[(long)rcv * 256 + c],       w0 * xs0 * y0);
    atomicAdd(&m0[(long)rcv * 256 + 128 + c], w3 * dot * inv_sqrt3);
    float oa = w1 * xs0;
    atomicAdd(&m1[(long)rcv * 768 + 0 * 256 + c], oa * y1a);
    atomicAdd(&m1[(long)rcv * 768 + 1 * 256 + c], oa * y1b);
    atomicAdd(&m1[(long)rcv * 768 + 2 * 256 + c], oa * y1c);
    float ob = w2 * y0;
    atomicAdd(&m1[(long)rcv * 768 + 0 * 256 + 128 + c], ob * xs1a);
    atomicAdd(&m1[(long)rcv * 768 + 1 * 256 + 128 + c], ob * xs1b);
    atomicAdd(&m1[(long)rcv * 768 + 2 * 256 + 128 + c], ob * xs1c);
  }
}

// ---------------------------------------------------------------------------
extern "C" void kernel_launch(void* const* d_in, const int* in_sizes, int n_in,
                              void* d_out, int out_size, void* d_ws, size_t ws_size,
                              hipStream_t stream) {
  const float* node_feats = (const float*)d_in[1];
  const float* edge_attrs = (const float*)d_in[2];
  const float* edge_feats = (const float*)d_in[3];
  const int*   edge_index = (const int*)d_in[4];
  const float* W_up0  = (const float*)d_in[5];
  const float* W_up1  = (const float*)d_in[6];
  const float* W_down = (const float*)d_in[7];
  const float* mlp_w1 = (const float*)d_in[8];
  const float* mlp_w2 = (const float*)d_in[9];
  const float* mlp_w3 = (const float*)d_in[10];
  const float* mlp_w4 = (const float*)d_in[11];
  const float* W_lin0 = (const float*)d_in[12];
  const float* W_lin1 = (const float*)d_in[13];
  const float* W_skip0 = (const float*)d_in[14];
  const float* W_skip1 = (const float*)d_in[15];

  const int N = in_sizes[1] / 512;   // node_feats (N, 4C)
  const int E = in_sizes[3] / 8;     // edge_feats (E, R)

  // Workspace layout
  char* ws = (char*)d_ws;
  size_t off = 0;
  auto alloc = [&](size_t bytes) -> char* {
    char* p = ws + off;
    off += (bytes + 255) & ~(size_t)255;
    return p;
  };
  unsigned short* up0p  = (unsigned short*)alloc((size_t)4 * 8 * 32 * 16 * 2);
  unsigned short* up1p  = (unsigned short*)alloc((size_t)4 * 8 * 32 * 16 * 2);
  unsigned short* downp = (unsigned short*)alloc((size_t)4 * 4 * 32 * 16 * 2);
  unsigned short* w1p   = (unsigned short*)alloc((size_t)5 * 16 * 32 * 16 * 2);
  unsigned short* w2p   = (unsigned short*)alloc((size_t)8 * 16 * 32 * 16 * 2);
  unsigned short* w3p   = (unsigned short*)alloc((size_t)8 * 16 * 32 * 16 * 2);
  unsigned short* w4p   = (unsigned short*)alloc((size_t)8 * 32 * 32 * 16 * 2);
  unsigned short* lin0p = (unsigned short*)alloc((size_t)8 * 8 * 32 * 16 * 2);
  unsigned short* lin1p = (unsigned short*)alloc((size_t)8 * 8 * 32 * 16 * 2);
  unsigned short* sk0p  = (unsigned short*)alloc((size_t)4 * 8 * 32 * 16 * 2);
  unsigned short* sk1p  = (unsigned short*)alloc((size_t)4 * 8 * 32 * 16 * 2);
  float* u0ws   = (float*)alloc((size_t)N * 128 * 4);
  float* u1ws   = (float*)alloc((size_t)N * 384 * 4);
  float* downws = (float*)alloc((size_t)N * 64 * 4);
  float* m0ws   = (float*)alloc((size_t)N * 256 * 4);
  float* m1ws   = (float*)alloc((size_t)N * 768 * 4);   // contiguous after m0ws

  const float inv_c  = 0.088388347648318447f;   // 1/sqrt(128)
  const float s_mlp1 = 0.085749292571254418f;   // 1/sqrt(136)
  const float s_256  = 0.0625f;                 // 1/sqrt(256)
  const float inv_avg = 1.0f / 16.0f;           // 1/AVG_NEIGH

  auto packGrid = [](int kt, int nt) { return dim3((kt * nt * 32 + 255) / 256); };
  // ---- pack weights (bf16, WMMA-B fragment layout, 1/sqrt(K) folded) ----
  pack_weights<<<packGrid(4, 8),  256, 0, stream>>>(W_up0,  128, 128, up0p,  4, 8,  inv_c);
  pack_weights<<<packGrid(4, 8),  256, 0, stream>>>(W_up1,  128, 128, up1p,  4, 8,  inv_c);
  pack_weights<<<packGrid(4, 4),  256, 0, stream>>>(W_down, 128, 64,  downp, 4, 4,  inv_c);
  pack_weights<<<packGrid(5, 16), 256, 0, stream>>>(mlp_w1, 136, 256, w1p,   5, 16, s_mlp1);
  pack_weights<<<packGrid(8, 16), 256, 0, stream>>>(mlp_w2, 256, 256, w2p,   8, 16, s_256);
  pack_weights<<<packGrid(8, 16), 256, 0, stream>>>(mlp_w3, 256, 256, w3p,   8, 16, s_256);
  pack_weights<<<packGrid(8, 32), 256, 0, stream>>>(mlp_w4, 256, 512, w4p,   8, 32, s_256);
  pack_weights<<<packGrid(8, 8),  256, 0, stream>>>(W_lin0, 256, 128, lin0p, 8, 8,  s_256);
  pack_weights<<<packGrid(8, 8),  256, 0, stream>>>(W_lin1, 256, 128, lin1p, 8, 8,  s_256);
  pack_weights<<<packGrid(4, 8),  256, 0, stream>>>(W_skip0, 128, 128, sk0p, 4, 8,  inv_c);
  pack_weights<<<packGrid(4, 8),  256, 0, stream>>>(W_skip1, 128, 128, sk1p, 4, 8,  inv_c);

  const int mtiles = N / 16;  // N = 10000 -> 625
  float* out_msg = (float*)d_out;                    // message (N,128,4)
  float* out_sc  = (float*)d_out + (size_t)N * 512;  // sc (N,512)

  // ---- node GEMMs ----
  wmma_gemm<4, 8><<<mtiles, 32, 0, stream>>>(node_feats, 512, 1, sk0p, out_sc, 512, 1, 1.0f);
  for (int i = 0; i < 3; ++i)
    wmma_gemm<4, 8><<<mtiles, 32, 0, stream>>>(node_feats + 128 + i, 512, 3, sk1p,
                                               out_sc + 128 + i, 512, 3, 1.0f);
  wmma_gemm<4, 8><<<mtiles, 32, 0, stream>>>(node_feats, 512, 1, up0p, u0ws, 128, 1, 1.0f);
  for (int i = 0; i < 3; ++i)
    wmma_gemm<4, 8><<<mtiles, 32, 0, stream>>>(node_feats + 128 + i, 512, 3, up1p,
                                               u1ws + i * 128, 384, 1, 1.0f);
  wmma_gemm<4, 4><<<mtiles, 32, 0, stream>>>(node_feats, 512, 1, downp, downws, 64, 1, 1.0f);

  // ---- zero message accumulators (m0 ++ m1 contiguous) ----
  zero_f32<<<2048, 256, 0, stream>>>(m0ws, (long)N * (256 + 768));

  // ---- fused edge MLP + tensor product + scatter (64 edges/block) ----
  const int edge_smem = 40960 + 32768 + 65536;  // 136 KB dynamic LDS
  edge_mlp_scatter<<<E / 64, 128, edge_smem, stream>>>(edge_feats, edge_attrs,
                                                       edge_index, E,
                                                       downws, u0ws, u1ws,
                                                       w1p, w2p, w3p, w4p,
                                                       m0ws, m1ws);

  // ---- final GEMMs, interleaved into message (N,128,4) ----
  wmma_gemm<8, 8><<<mtiles, 32, 0, stream>>>(m0ws, 256, 1, lin0p, out_msg + 0, 512, 4, inv_avg);
  for (int i = 0; i < 3; ++i)
    wmma_gemm<8, 8><<<mtiles, 32, 0, stream>>>(m1ws + i * 256, 768, 1, lin1p,
                                               out_msg + 1 + i, 512, 4, inv_avg);
}
